// SO3Grid_36962488549536
// MI455X (gfx1250) — matrix operand order repaired and checked by reading, not compiled, not run on hardware
//
#include <hip/hip_runtime.h>

// ---------------------------------------------------------------------------
// out = features @ M,  M[j,i] = sum_r D[r,j] * w[r] * D[r,i],
// w[r] = qw[(r/NA) % NB]   (the 1/sqrt(dim) and sqrt(dim) scales cancel).
//
// Fast path (needs ~154MB workspace):
//   1) transpose_kernel: Dt[col][r]=bf16(D[r][col]), Dtw=qw-weighted copy.
//      Both copies (152MB) fit in the 192MB L2 together.
//   2) gram_async_kernel: M = Dtw^T-layout @ Dt via bf16 WMMA, staging tiles
//      with global_load_async_to_lds_b128 (ASYNCcnt) + double-buffered LDS.
//   3) apply_kernel: out = features @ M (bf16 WMMA).
// Fallback path (small ws): direct-from-f32 gram kernel (round-1 style,
// integer divisions hoisted to uniform scalars) + apply_kernel.
// ---------------------------------------------------------------------------

typedef __attribute__((ext_vector_type(16))) __bf16 v16bf;
typedef __attribute__((ext_vector_type(8)))  __bf16 v8bf;
typedef __attribute__((ext_vector_type(8)))  float  v8f;

#define BATCH   1024
#define NA      48
#define NB      24
#define DIM     680
#define MPAD    688                 // padded leading dim (43*16)
#define RTOT    (NA * NB * NA)      // 55296 contraction length
#define KCHUNKS (RTOT / 32)         // 1728
#define KSLICES 8
#define KPER    (KCHUNKS / KSLICES) // 216
#define LDK     48                  // LDS k-stride (halves) for f32-staged path
#define LDA     40                  // LDS k-stride (halves) for async path (80B rows)

__device__ __forceinline__ __bf16 tobf(float f) { return (__bf16)f; }

// Low 32 bits of a generic pointer to __shared__ == LDS byte offset (flat
// aperture truncation per CDNA5 ISA 10.2).
__device__ __forceinline__ unsigned lds_addr32(const void* p) {
    return (unsigned)(uintptr_t)p;
}

// Async copy 16B global -> LDS, tracked by ASYNCcnt.
__device__ __forceinline__ void async_b128(const void* gsrc, void* ldst) {
    asm volatile("global_load_async_to_lds_b128 %0, %1, off"
                 :: "v"(lds_addr32(ldst)), "v"(gsrc)
                 : "memory");
}
__device__ __forceinline__ void wait_async_le4() {
    asm volatile("s_wait_asynccnt 0x4" ::: "memory");
}

// A fragment (16x32 bf16 tile at row jb) from LDS [row][k], stride ldk halves.
template <int ldk>
__device__ __forceinline__ v16bf load_a_frag(const __bf16* s, int jb, int lane) {
    const int row = jb + (lane & 15);
    const int kb  = (lane >> 4) * 16;             // 0 or 16
    const __bf16* p = &s[row * ldk + kb];
    v8bf lo = *(const v8bf*)p;                    // K = kb .. kb+7
    v8bf hi = *(const v8bf*)(p + 8);              // K = kb+8 .. kb+15
    v16bf a;
#pragma unroll
    for (int t = 0; t < 8; ++t) { a[t] = lo[t]; a[t + 8] = hi[t]; }
    return a;
}

// B fragment (32x16 bf16 tile at col ib) from LDS [col][k], stride ldk halves.
template <int ldk>
__device__ __forceinline__ v16bf load_b_frag(const __bf16* s, int ib, int lane) {
    const int col = ib + (lane & 15);
    const int kb  = (lane >> 4) * 16;             // 0 or 16
    return *(const v16bf*)&s[col * ldk + kb];
}

#define WMMA_BF16(A, B, C) \
    __builtin_amdgcn_wmma_f32_16x16x32_bf16(false, (A), false, (B), (short)0, (C), false, false)

// ---------------------------------------------------------------------------
// Pre-pass: Dt[col][r] = bf16(D[r][col]); Dtw[col][r] = bf16(D[r][col]*w[r]).
// 64x64 tiles through LDS so both global accesses are coalesced.
// ---------------------------------------------------------------------------
__global__ __launch_bounds__(256) void transpose_kernel(const float* __restrict__ D,
                                                        const float* __restrict__ qw,
                                                        __bf16* __restrict__ Dt,
                                                        __bf16* __restrict__ Dtw) {
    __shared__ float ts[64 * 65];
    const int tid = threadIdx.x;
    const int r0  = blockIdx.x * 64;   // 864 tiles cover RTOT exactly
    const int c0  = blockIdx.y * 64;   // 11 tiles cover MPAD (guarded)

#pragma unroll
    for (int e = tid; e < 64 * 64; e += 256) {
        const int rr = e >> 6, cc = e & 63;
        const int col = c0 + cc;
        const float v = (col < DIM) ? D[(size_t)(r0 + rr) * DIM + col] : 0.0f;
        ts[rr * 65 + cc] = v;
    }
    __syncthreads();
#pragma unroll
    for (int e = tid; e < 64 * 64; e += 256) {
        const int cc = e >> 6, rr = e & 63;
        const int col = c0 + cc;
        if (col < MPAD) {
            const int r = r0 + rr;
            const float v = ts[rr * 65 + cc];
            const float w = qw[(r / NA) % NB];       // one-time pass: div OK
            Dt [(size_t)col * RTOT + r] = tobf(v);
            Dtw[(size_t)col * RTOT + r] = tobf(v * w);
        }
    }
}

// ---------------------------------------------------------------------------
// Fast gram kernel: 128(j) x 128(i) block tile, 8 waves as 2x4 (64x32 each,
// 2x2x... = 4x2 = 8 WMMA accumulators). Tiles staged with async-to-LDS B128
// into a double buffer; operands come pre-transposed & pre-converted.
// ---------------------------------------------------------------------------
__global__ __launch_bounds__(256) void gram_async_kernel(const __bf16* __restrict__ Dt,
                                                         const __bf16* __restrict__ Dtw,
                                                         float* __restrict__ M) {
    // Buffer layout: rows 0..127 = A (j cols of Dtw), rows 128..255 = B (i cols of Dt)
    __shared__ __align__(16) __bf16 sAB[2][256 * LDA];

    const int tid  = threadIdx.x;
    const int lane = tid & 31;
    const int wave = tid >> 5;
    const int wj   = wave >> 2;   // 0..1 -> j offset 64*wj
    const int wi   = wave & 3;    // 0..3 -> i offset 32*wi

    const int i0 = blockIdx.x * 128;
    const int j0 = blockIdx.y * 128;
    const int z  = blockIdx.z;

    v8f c[4][2];
#pragma unroll
    for (int a = 0; a < 4; ++a)
#pragma unroll
        for (int b = 0; b < 2; ++b) c[a][b] = (v8f){};

    // Issue one 256-row x 64B tile: 1024 B128 ops, 4 per thread.
    auto issue_tile = [&](int buf, int step) {
        const int r0 = (z * KPER + step) * 32;
#pragma unroll
        for (int q = 0; q < 4; ++q) {
            const int idx   = tid + 256 * q;     // 0..1023
            const int row   = idx >> 2;          // 0..255
            const int chunk = idx & 3;           // 16B chunk within 64B row
            const bool isA  = row < 128;
            int col = isA ? (j0 + row) : (i0 + (row - 128));
            col = (col < MPAD) ? col : (MPAD - 1);      // clamp: garbage is store-guarded
            const __bf16* g = (isA ? Dtw : Dt) + (size_t)col * RTOT + r0 + chunk * 8;
            async_b128(g, &sAB[buf][row * LDA + chunk * 8]);
        }
    };

    issue_tile(0, 0);
    for (int s = 0; s < KPER; ++s) {
        const int nxt = (s + 1 < KPER) ? (s + 1) : (KPER - 1);
        issue_tile((s + 1) & 1, nxt);     // prefetch next tile
        wait_async_le4();                  // oldest 4 (current tile) complete
        __syncthreads();                   // all waves' copies visible

        const __bf16* base = sAB[s & 1];
        const __bf16* bA = base;                 // rows 0..127 : A
        const __bf16* bB = base + 128 * LDA;     // rows 128..255: B

        v16bf af[4], bf_[2];
#pragma unroll
        for (int a = 0; a < 4; ++a) af[a]  = load_a_frag<LDA>(bA, wj * 64 + 16 * a, lane);
#pragma unroll
        for (int b = 0; b < 2; ++b) bf_[b] = load_b_frag<LDA>(bB, wi * 32 + 16 * b, lane);

#pragma unroll
        for (int a = 0; a < 4; ++a)
#pragma unroll
            for (int b = 0; b < 2; ++b)
                c[a][b] = WMMA_BF16(af[a], bf_[b], c[a][b]);

        __syncthreads();                   // protect buffer before next prefetch lands
    }

    // Accumulate into M. C layout: m = g + 8*(lane>=16), n = lane&15.
    const int m_off = (lane >> 4) * 8;
    const int n_off = lane & 15;
#pragma unroll
    for (int a = 0; a < 4; ++a)
#pragma unroll
        for (int b = 0; b < 2; ++b)
#pragma unroll
            for (int g = 0; g < 8; ++g) {
                const int j = j0 + wj * 64 + 16 * a + m_off + g;
                const int i = i0 + wi * 32 + 16 * b + n_off;
                if (j < DIM && i < DIM) atomicAdd(&M[j * MPAD + i], c[a][b][g]);
            }
}

// ---------------------------------------------------------------------------
// Fallback gram kernel (small ws): direct from f32 D, 128(j) x 64(i) tile,
// uniform-hoisted qw weights (no per-element division).
// ---------------------------------------------------------------------------
__global__ __launch_bounds__(256) void gram_fallback_kernel(const float* __restrict__ D,
                                                            const float* __restrict__ qw,
                                                            float* __restrict__ M) {
    __shared__ __align__(32) __bf16 sA[128 * LDK];
    __shared__ __align__(32) __bf16 sB[64 * LDK];

    const int tid  = threadIdx.x;
    const int lane = tid & 31;
    const int wave = tid >> 5;
    const int wj   = wave >> 1;
    const int wi   = wave & 1;

    const int i0 = blockIdx.x * 64;
    const int j0 = blockIdx.y * 128;
    const int z  = blockIdx.z;

    v8f c00 = {}, c01 = {}, c10 = {}, c11 = {};

    for (int step = 0; step < KPER; ++step) {
        const int r0 = (z * KPER + step) * 32;
        // qw lookup hoisted: a 32-chunk crosses at most one alpha boundary.
        const int   q0  = r0 / NA;                 // uniform
        const float wa  = qw[q0 % NB];
        const float wb  = qw[(q0 + 1) % NB];
        const int   cut = NA - (r0 - q0 * NA);     // k < cut -> wa

#pragma unroll
        for (int e = tid; e < 32 * 128; e += 256) {
            const int k = e >> 7, j = e & 127;
            const int col = j0 + j;
            const float v = (col < DIM) ? D[(size_t)(r0 + k) * DIM + col] : 0.0f;
            const float w = (k < cut) ? wa : wb;
            sA[j * LDK + k] = tobf(v * w);
        }
#pragma unroll
        for (int e = tid; e < 32 * 64; e += 256) {
            const int k = e >> 6, i = e & 63;
            const int col = i0 + i;
            const float v = (col < DIM) ? D[(size_t)(r0 + k) * DIM + col] : 0.0f;
            sB[i * LDK + k] = tobf(v);
        }
        __syncthreads();

        const v16bf a0 = load_a_frag<LDK>(sA, wj * 32 +  0, lane);
        const v16bf a1 = load_a_frag<LDK>(sA, wj * 32 + 16, lane);
        const v16bf b0 = load_b_frag<LDK>(sB, wi * 32 +  0, lane);
        const v16bf b1 = load_b_frag<LDK>(sB, wi * 32 + 16, lane);

        c00 = WMMA_BF16(a0, b0, c00);
        c01 = WMMA_BF16(a0, b1, c01);
        c10 = WMMA_BF16(a1, b0, c10);
        c11 = WMMA_BF16(a1, b1, c11);

        __syncthreads();
    }

    const int m_off = (lane >> 4) * 8;
    const int n_off = lane & 15;
#pragma unroll
    for (int g = 0; g < 8; ++g) {
        const int jA = j0 + wj * 32 + m_off + g, jB = jA + 16;
        const int iA = i0 + wi * 32 + n_off,     iB = iA + 16;
        if (jA < DIM && iA < DIM) atomicAdd(&M[jA * MPAD + iA], c00[g]);
        if (jA < DIM && iB < DIM) atomicAdd(&M[jA * MPAD + iB], c01[g]);
        if (jB < DIM && iA < DIM) atomicAdd(&M[jB * MPAD + iA], c10[g]);
        if (jB < DIM && iB < DIM) atomicAdd(&M[jB * MPAD + iB], c11[g]);
    }
}

// ---------------------------------------------------------------------------
// out[n,i] = sum_j features[n,j] * M[j,i].  128(n) x 64(i) tile.
// ---------------------------------------------------------------------------
__global__ __launch_bounds__(256) void apply_kernel(const float* __restrict__ F,
                                                    const float* __restrict__ M,
                                                    float* __restrict__ out) {
    __shared__ __align__(32) __bf16 sA[128 * LDK];
    __shared__ __align__(32) __bf16 sB[64 * LDK];

    const int tid  = threadIdx.x;
    const int lane = tid & 31;
    const int wave = tid >> 5;
    const int wj   = wave >> 1;
    const int wi   = wave & 1;

    const int i0 = blockIdx.x * 64;
    const int n0 = blockIdx.y * 128;

    v8f c00 = {}, c01 = {}, c10 = {}, c11 = {};

    const int ksteps = (DIM + 31) / 32;   // 22
    for (int step = 0; step < ksteps; ++step) {
        const int r0 = step * 32;
#pragma unroll
        for (int e = tid; e < 128 * 32; e += 256) {
            const int nl = e >> 5, k = e & 31;
            const int r = r0 + k;
            const float v = (r < DIM) ? F[(size_t)(n0 + nl) * DIM + r] : 0.0f;
            sA[nl * LDK + k] = tobf(v);
        }
#pragma unroll
        for (int e = tid; e < 32 * 64; e += 256) {
            const int k = e >> 6, i = e & 63;
            const int r = r0 + k;
            const int col = i0 + i;
            const float v = (r < DIM && col < DIM) ? M[(size_t)r * MPAD + col] : 0.0f;
            sB[i * LDK + k] = tobf(v);
        }
        __syncthreads();

        const v16bf a0 = load_a_frag<LDK>(sA, wj * 32 +  0, lane);
        const v16bf a1 = load_a_frag<LDK>(sA, wj * 32 + 16, lane);
        const v16bf b0 = load_b_frag<LDK>(sB, wi * 32 +  0, lane);
        const v16bf b1 = load_b_frag<LDK>(sB, wi * 32 + 16, lane);

        c00 = WMMA_BF16(a0, b0, c00);
        c01 = WMMA_BF16(a0, b1, c01);
        c10 = WMMA_BF16(a1, b0, c10);
        c11 = WMMA_BF16(a1, b1, c11);

        __syncthreads();
    }

    const int m_off = (lane >> 4) * 8;
    const int n_off = lane & 15;
#pragma unroll
    for (int g = 0; g < 8; ++g) {
        const int rowA = n0 + wj * 32 + m_off + g, rowB = rowA + 16;
        const int colA = i0 + wi * 32 + n_off,     colB = colA + 16;
        if (colA < DIM) out[(size_t)rowA * DIM + colA] = c00[g];
        if (colB < DIM) out[(size_t)rowA * DIM + colB] = c01[g];
        if (colA < DIM) out[(size_t)rowB * DIM + colA] = c10[g];
        if (colB < DIM) out[(size_t)rowB * DIM + colB] = c11[g];
    }
}

// ---------------------------------------------------------------------------
extern "C" void kernel_launch(void* const* d_in, const int* in_sizes, int n_in,
                              void* d_out, int out_size, void* d_ws, size_t ws_size,
                              hipStream_t stream) {
    const float* features = (const float*)d_in[0];
    const float* D        = (const float*)d_in[1];
    const float* qw       = (const float*)d_in[2];
    float*       out      = (float*)d_out;

    const size_t bytes_copy = (size_t)MPAD * RTOT * sizeof(__bf16);  // 76.1 MB
    const size_t bytes_M    = (size_t)MPAD * MPAD * sizeof(float);   // 1.9 MB
    const size_t need_fast  = 2 * bytes_copy + bytes_M;

    float* M;
    if (ws_size >= need_fast) {
        __bf16* Dt  = (__bf16*)d_ws;
        __bf16* Dtw = (__bf16*)((char*)d_ws + bytes_copy);
        M           = (float*)((char*)d_ws + 2 * bytes_copy);

        hipMemsetAsync(M, 0, bytes_M, stream);

        dim3 gt(RTOT / 64, (MPAD + 63) / 64, 1);                    // (864, 11)
        transpose_kernel<<<gt, 256, 0, stream>>>(D, qw, Dt, Dtw);

        dim3 g1((MPAD + 127) / 128, (MPAD + 127) / 128, KSLICES);   // (6, 6, 8)
        gram_async_kernel<<<g1, 256, 0, stream>>>(Dt, Dtw, M);
    } else {
        M = (float*)d_ws;
        hipMemsetAsync(M, 0, bytes_M, stream);

        dim3 g1((MPAD + 63) / 64, (MPAD + 127) / 128, KSLICES);     // (11, 6, 8)
        gram_fallback_kernel<<<g1, 256, 0, stream>>>(D, qw, M);
    }

    dim3 g2((DIM + 63) / 64, BATCH / 128, 1);                       // (11, 8)
    apply_kernel<<<g2, 256, 0, stream>>>(features, M, out);
}
